// MultiHeadAttention_37400575213828
// MI455X (gfx1250) — matrix operand order repaired and checked by reading, not compiled, not run on hardware
//
#include <hip/hip_runtime.h>
#include <hip/hip_bf16.h>

#define B_  4
#define S_  2048
#define E_  512
#define H_  8
#define D_  512
#define DM_ 512

typedef __attribute__((ext_vector_type(16))) __bf16 v16bf;
typedef __attribute__((ext_vector_type(8)))  float  v8f;

static __device__ __forceinline__ unsigned short f32_to_bf16(float f) {
  unsigned int u = __float_as_uint(f);
  u += 0x7FFFu + ((u >> 16) & 1u);          // round to nearest even
  return (unsigned short)(u >> 16);
}

// ---------------------------------------------------------------------------
// f32 -> bf16 conversion (grid-stride)
// ---------------------------------------------------------------------------
__global__ void cvt_f32_to_bf16(const float* __restrict__ src,
                                unsigned short* __restrict__ dst, int n) {
  int i = blockIdx.x * blockDim.x + threadIdx.x;
  const int stride = gridDim.x * blockDim.x;
  for (; i < n; i += stride) dst[i] = f32_to_bf16(src[i]);
}

// ---------------------------------------------------------------------------
// Vectorized WMMA fragment loaders (bf16 16x16x32), ds_load_b128 only.
// A 16x32 (MxK), row-major [M][K]: lane's data = two contiguous 8-elem runs:
//   elems 0..7  = K[k0 + half*8 .. +7],  elems 8..15 = K[k0 + 16 + half*8 .. +7]
// ---------------------------------------------------------------------------
static __device__ __forceinline__ v16bf load_a_frag_v(const unsigned short* sm, int ld,
                                                      int row0, int k0, int lane) {
  const int m = row0 + (lane & 15);
  const int half = lane >> 4;
  union { uint4 q[2]; v16bf v; } u;
  u.q[0] = *(const uint4*)(sm + m * ld + k0 + half * 8);
  u.q[1] = *(const uint4*)(sm + m * ld + k0 + 16 + half * 8);
  return u.v;
}
// B 32x16 (KxN) with LDS holding B **K-contiguous per column**: sm[n*ldk + k].
// lane's data = two 8-elem runs: K[k0 + half*8 .. +7], K[k0 + 16 + half*8 .. +7]
static __device__ __forceinline__ v16bf load_bt_frag_v(const unsigned short* sm, int ldk,
                                                       int k0, int col0, int lane) {
  const int n = col0 + (lane & 15);
  const int half = lane >> 4;
  union { uint4 q[2]; v16bf v; } u;
  const uint4* p = (const uint4*)(sm + n * ldk + k0 + half * 8);
  u.q[0] = p[0];
  u.q[1] = p[2];
  return u.v;
}

// ---------------------------------------------------------------------------
// Tiled GEMM: C[z] = (A[b] @ W[h] + bias[h]) * alpha
//   A: [M,K] bf16 row-major, W: [K,N] bf16 row-major, bias: [N] f32
//   block tile 128x128, 8 waves in 4(M) x 2(N), wave tile 32x64 = 2x4 frags
//   W tile staged transposed (BsT[n][k]) so B-frag loads are ds_load_b128.
// ---------------------------------------------------------------------------
template <bool OUT_F32>
__global__ void __launch_bounds__(256)
gemm_bias_bf16(const unsigned short* __restrict__ A0, long long strideAb,
               const unsigned short* __restrict__ W0, long long strideWh,
               const float* __restrict__ bias0, long long strideBh,
               void* __restrict__ C0, long long strideCz,
               int M, int N, int K, int Hdiv, float alpha) {
  __shared__ __align__(16) unsigned short As[128 * 32];   // [m][k]
  __shared__ __align__(16) unsigned short BsT[128 * 32];  // [n][k]
  const int z = blockIdx.z;
  const int b = z / Hdiv, h = z % Hdiv;
  const unsigned short* A = A0 + (long long)b * strideAb;
  const unsigned short* W = W0 + (long long)h * strideWh;
  const float* bias = bias0 + (long long)h * strideBh;
  const int n0 = blockIdx.x * 128;
  const int m0 = blockIdx.y * 128;
  const int tid = threadIdx.x;
  const int lane = tid & 31;
  const int wid = tid >> 5;
  const int wm = wid >> 1;   // 0..3
  const int wn = wid & 1;    // 0..1

  v8f acc[2][4];
#pragma unroll
  for (int im = 0; im < 2; ++im)
#pragma unroll
    for (int in = 0; in < 4; ++in)
#pragma unroll
      for (int j = 0; j < 8; ++j) acc[im][in][j] = 0.0f;

  for (int k0 = 0; k0 < K; k0 += 32) {
    {  // stage A tile (128 rows x 32 k): two b128 per thread
      const int r = tid >> 1;
      const int hc = (tid & 1) * 16;
      const uint4* src = (const uint4*)(A + (size_t)(m0 + r) * K + k0 + hc);
      uint4* dst = (uint4*)(As + r * 32 + hc);
      dst[0] = src[0];
      dst[1] = src[1];
    }
    {  // stage W tile transposed: thread owns a 2(k) x 8(n) subtile
      const int kk = (tid >> 4) * 2;     // 0,2,..,30
      const int ng = (tid & 15) * 8;     // 0,8,..,120
      union { uint4 q; unsigned short hh[8]; } r0, r1;
      r0.q = *(const uint4*)(W + (size_t)(k0 + kk) * N + n0 + ng);
      r1.q = *(const uint4*)(W + (size_t)(k0 + kk + 1) * N + n0 + ng);
#pragma unroll
      for (int cc = 0; cc < 8; ++cc) {
        const unsigned int dw = (unsigned int)r0.hh[cc] | ((unsigned int)r1.hh[cc] << 16);
        *(unsigned int*)(BsT + (ng + cc) * 32 + kk) = dw;
      }
    }
    __syncthreads();
    v16bf af[2], bfr[4];
#pragma unroll
    for (int im = 0; im < 2; ++im) af[im] = load_a_frag_v(As, 32, wm * 32 + im * 16, 0, lane);
#pragma unroll
    for (int in = 0; in < 4; ++in) bfr[in] = load_bt_frag_v(BsT, 32, 0, wn * 64 + in * 16, lane);
#pragma unroll
    for (int im = 0; im < 2; ++im)
#pragma unroll
      for (int in = 0; in < 4; ++in)
        acc[im][in] = __builtin_amdgcn_wmma_f32_16x16x32_bf16(
            false, af[im], false, bfr[in], (short)0, acc[im][in], false, false);
    __syncthreads();
  }

  const int half = lane >> 4;
#pragma unroll
  for (int im = 0; im < 2; ++im) {
#pragma unroll
    for (int in = 0; in < 4; ++in) {
      const int col = n0 + wn * 64 + in * 16 + (lane & 15);
      const float bc = bias[col];
#pragma unroll
      for (int j = 0; j < 8; ++j) {
        const int row = m0 + wm * 32 + im * 16 + j + 8 * half;
        const float v = (acc[im][in][j] + bc) * alpha;
        if (OUT_F32) {
          float* C = (float*)C0 + (long long)z * strideCz;
          C[(size_t)row * N + col] = v;
        } else {
          unsigned short* C = (unsigned short*)C0 + (long long)z * strideCz;
          C[(size_t)row * N + col] = f32_to_bf16(v);
        }
      }
    }
  }
}

// ---------------------------------------------------------------------------
// Flash attention: per (b,h), 16 query rows per block, 32-key tiles.
// Q pre-scaled by 1/sqrt(D). Online softmax, ctx accumulated in f32 frags.
// QK^T split across waves by D-slice; partials merged with ds_add_f32.
// KVs buffer holds K row-major [t][d] (already B^T layout for QK^T), then is
// overwritten by V transposed (Vt[d][t]) so PV B-frags are also b128 loads.
// ---------------------------------------------------------------------------
__global__ void __launch_bounds__(256)
flash_attn_kernel(const unsigned short* __restrict__ Qg,
                  const unsigned short* __restrict__ Kg,
                  const unsigned short* __restrict__ Vg,
                  unsigned short* __restrict__ Ctx,
                  int Sdim, int Ddim, int Hn) {
  __shared__ __align__(16) unsigned short Qs[16 * 512];   // 16 KB [row][d]
  __shared__ __align__(16) unsigned short KVs[32 * 512];  // 32 KB K[t][d] -> Vt[d][t]
  __shared__ __align__(16) unsigned short Ps[16 * 32];    // 1 KB  P bf16 [row][t]
  __shared__ __align__(16) float Sred[16 * 32];           // 2 KB  logits f32 [row][t]
  const int z = blockIdx.y;                               // b*H + h
  const size_t base = (size_t)z * Sdim * Ddim;
  const int s0 = blockIdx.x * 16;
  const int tid = threadIdx.x;
  const int lane = tid & 31;
  const int wid = tid >> 5;
  const int half = lane >> 4;

  // stage Q tile (16 x 512)
#pragma unroll
  for (int c = 0; c < 2; ++c) {
    const int id = tid + c * 256;          // 0..511
    const int r = id >> 5;                 // 0..15
    const int off = (id & 31) * 16;
    const uint4* src = (const uint4*)(Qg + base + (size_t)(s0 + r) * Ddim + off);
    uint4* dst = (uint4*)(Qs + r * 512 + off);
    dst[0] = src[0];
    dst[1] = src[1];
  }

  float m_st[8], l_st[8];
  v8f acc[4];
#pragma unroll
  for (int j = 0; j < 8; ++j) { m_st[j] = -3.0e38f; l_st[j] = 0.0f; }
#pragma unroll
  for (int in = 0; in < 4; ++in)
#pragma unroll
    for (int j = 0; j < 8; ++j) acc[in][j] = 0.0f;

  const int dslice = Ddim >> 3;            // 64: D-range reduced per wave
  for (int t0 = 0; t0 < Sdim; t0 += 32) {
    __syncthreads();  // previous-tile reads of KVs/Ps/Sred complete
    // stage K tile (32 x 512) row-major; zero the logit reduction tile
#pragma unroll
    for (int c = 0; c < 4; ++c) {
      const int id = tid + c * 256;        // 0..1023
      const int r = id >> 5;               // 0..31
      const int off = (id & 31) * 16;
      const uint4* src = (const uint4*)(Kg + base + (size_t)(t0 + r) * Ddim + off);
      uint4* dst = (uint4*)(KVs + r * 512 + off);
      dst[0] = src[0];
      dst[1] = src[1];
    }
    *(float2*)(Sred + tid * 2) = make_float2(0.0f, 0.0f);
    __syncthreads();

    // partial S = Q @ K^T over this wave's D-slice (4 WMMA), merge via ds_add_f32
    {
      v8f sacc[2];
#pragma unroll
      for (int in = 0; in < 2; ++in)
#pragma unroll
        for (int j = 0; j < 8; ++j) sacc[in][j] = 0.0f;
      const int dw0 = wid * dslice;
#pragma unroll
      for (int dd = 0; dd < 2; ++dd) {
        const int d0 = dw0 + dd * 32;
        const v16bf aq = load_a_frag_v(Qs, 512, 0, d0, lane);
#pragma unroll
        for (int in = 0; in < 2; ++in) {
          const v16bf bk = load_bt_frag_v(KVs, 512, d0, in * 16, lane);
          sacc[in] = __builtin_amdgcn_wmma_f32_16x16x32_bf16(
              false, aq, false, bk, (short)0, sacc[in], false, false);
        }
      }
#pragma unroll
      for (int in = 0; in < 2; ++in)
#pragma unroll
        for (int j = 0; j < 8; ++j)
          atomicAdd(&Sred[(j + 8 * half) * 32 + in * 16 + (lane & 15)], sacc[in][j]);
    }
    __syncthreads();

    // online softmax: all waves read the full 16x32 tile (identical math)
    {
      float sv0[8], sv1[8], p0[8], p1[8];
#pragma unroll
      for (int j = 0; j < 8; ++j) {
        sv0[j] = Sred[(j + 8 * half) * 32 + (lane & 15)];
        sv1[j] = Sred[(j + 8 * half) * 32 + 16 + (lane & 15)];
      }
#pragma unroll
      for (int j = 0; j < 8; ++j) {
        float t = fmaxf(sv0[j], sv1[j]);
        t = fmaxf(t, __shfl_xor(t, 1, 32));
        t = fmaxf(t, __shfl_xor(t, 2, 32));
        t = fmaxf(t, __shfl_xor(t, 4, 32));
        t = fmaxf(t, __shfl_xor(t, 8, 32));
        const float mn = fmaxf(m_st[j], t);
        const float sc = __expf(m_st[j] - mn);
        m_st[j] = mn;
        p0[j] = __expf(sv0[j] - mn);
        p1[j] = __expf(sv1[j] - mn);
        float s = p0[j] + p1[j];
        s += __shfl_xor(s, 1, 32);
        s += __shfl_xor(s, 2, 32);
        s += __shfl_xor(s, 4, 32);
        s += __shfl_xor(s, 8, 32);
        l_st[j] = l_st[j] * sc + s;
#pragma unroll
        for (int in = 0; in < 4; ++in) acc[in][j] *= sc;
      }
      if (wid == 0) {  // one wave publishes P (bf16)
#pragma unroll
        for (int j = 0; j < 8; ++j) {
          const int r = j + 8 * half;
          Ps[r * 32 + (lane & 15)]      = f32_to_bf16(p0[j]);
          Ps[r * 32 + 16 + (lane & 15)] = f32_to_bf16(p1[j]);
        }
      }
    }
    __syncthreads();  // K/Sred reads + Ps writes done

    // stage V tile transposed: Vt[d][t], thread owns 4 subtiles of 2(t) x 8(d)
#pragma unroll
    for (int c = 0; c < 4; ++c) {
      const int id = tid + c * 256;        // 0..1023
      const int d0r = (id & 63) * 8;       // 0..504
      const int tt = (id >> 6) * 2;        // 0,2,..,30
      union { uint4 q; unsigned short hh[8]; } r0, r1;
      r0.q = *(const uint4*)(Vg + base + (size_t)(t0 + tt) * Ddim + d0r);
      r1.q = *(const uint4*)(Vg + base + (size_t)(t0 + tt + 1) * Ddim + d0r);
#pragma unroll
      for (int cc = 0; cc < 8; ++cc) {
        const unsigned int dw = (unsigned int)r0.hh[cc] | ((unsigned int)r1.hh[cc] << 16);
        *(unsigned int*)(KVs + (d0r + cc) * 32 + tt) = dw;
      }
    }
    if (t0 + 32 < Sdim) {  // prefetch next K tile -> global_prefetch_b8
      __builtin_prefetch(Kg + base + (size_t)(t0 + 32) * Ddim + tid * 64, 0, 1);
    }
    __syncthreads();

    // ctx += P @ V  (K dim = 32: one WMMA per output fragment)
    const v16bf pa = load_a_frag_v(Ps, 32, 0, 0, lane);
#pragma unroll
    for (int in = 0; in < 4; ++in) {
      const v16bf bv = load_bt_frag_v(KVs, 32, 0, wid * 64 + in * 16, lane);
      acc[in] = __builtin_amdgcn_wmma_f32_16x16x32_bf16(
          false, pa, false, bv, (short)0, acc[in], false, false);
    }
  }

  // finalize: ctx / l, store as [B, S, H*D] bf16
  const int bb = z / Hn, hh = z % Hn;
  const int HD = Hn * Ddim;
#pragma unroll
  for (int j = 0; j < 8; ++j) {
    const float inv = 1.0f / l_st[j];
    const int s = s0 + j + 8 * half;
#pragma unroll
    for (int in = 0; in < 4; ++in) {
      const int col = hh * Ddim + wid * 64 + in * 16 + (lane & 15);
      Ctx[((size_t)bb * Sdim + s) * HD + col] = f32_to_bf16(acc[in][j] * inv);
    }
  }
}

// ---------------------------------------------------------------------------
// Orchestration
// ---------------------------------------------------------------------------
extern "C" void kernel_launch(void* const* d_in, const int* in_sizes, int n_in,
                              void* d_out, int out_size, void* d_ws, size_t ws_size,
                              hipStream_t stream) {
  (void)in_sizes; (void)n_in; (void)out_size; (void)ws_size;
  const float* x  = (const float*)d_in[0];
  // d_in[1] = attention_mask (unused by the reference math)
  const float* Wq = (const float*)d_in[2];
  const float* bq = (const float*)d_in[3];
  const float* Wk = (const float*)d_in[4];
  const float* bk = (const float*)d_in[5];
  const float* Wv = (const float*)d_in[6];
  const float* bv = (const float*)d_in[7];
  const float* Wo = (const float*)d_in[8];
  const float* bo = (const float*)d_in[9];
  float* out = (float*)d_out;

  const size_t nx   = (size_t)B_ * S_ * E_;        // 4,194,304
  const size_t nw   = (size_t)H_ * E_ * D_;        // 2,097,152 (== H*D*DM for Wo)
  const size_t nqkv = (size_t)B_ * H_ * S_ * D_;   // 33,554,432

  unsigned short* ws  = (unsigned short*)d_ws;
  unsigned short* xb  = ws;
  unsigned short* wqb = xb + nx;
  unsigned short* wkb = wqb + nw;
  unsigned short* wvb = wkb + nw;
  unsigned short* wob = wvb + nw;
  unsigned short* Qw  = wob + nw;
  unsigned short* Kw  = Qw + nqkv;
  unsigned short* Vw  = Kw + nqkv;
  unsigned short* Cw  = Vw + nqkv;

  cvt_f32_to_bf16<<<1024, 256, 0, stream>>>(x,  xb,  (int)nx);
  cvt_f32_to_bf16<<<512,  256, 0, stream>>>(Wq, wqb, (int)nw);
  cvt_f32_to_bf16<<<512,  256, 0, stream>>>(Wk, wkb, (int)nw);
  cvt_f32_to_bf16<<<512,  256, 0, stream>>>(Wv, wvb, (int)nw);
  cvt_f32_to_bf16<<<512,  256, 0, stream>>>(Wo, wob, (int)nw);

  const float qscale = 0.044194173824159216f;  // 1/sqrt(512), folded into Q
  dim3 gqkv(D_ / 128, S_ / 128, B_ * H_);
  gemm_bias_bf16<false><<<gqkv, 256, 0, stream>>>(
      xb, (long long)S_ * E_, wqb, (long long)E_ * D_, bq, (long long)D_,
      Qw, (long long)S_ * D_, S_, D_, E_, H_, qscale);
  gemm_bias_bf16<false><<<gqkv, 256, 0, stream>>>(
      xb, (long long)S_ * E_, wkb, (long long)E_ * D_, bk, (long long)D_,
      Kw, (long long)S_ * D_, S_, D_, E_, H_, 1.0f);
  gemm_bias_bf16<false><<<gqkv, 256, 0, stream>>>(
      xb, (long long)S_ * E_, wvb, (long long)E_ * D_, bv, (long long)D_,
      Vw, (long long)S_ * D_, S_, D_, E_, H_, 1.0f);

  dim3 gfa(S_ / 16, B_ * H_);
  flash_attn_kernel<<<gfa, 256, 0, stream>>>(Qw, Kw, Vw, Cw, S_, D_, H_);

  dim3 gout(DM_ / 128, (B_ * S_) / 128, 1);
  gemm_bias_bf16<true><<<gout, 256, 0, stream>>>(
      Cw, 0LL, wob, 0LL, bo, 0LL, out, 0LL,
      B_ * S_, DM_, H_ * D_, 1, 1.0f);
}
